// _fasterRCNN_35940286333381
// MI455X (gfx1250) — matrix operand order repaired and checked by reading, not compiled, not run on hardware
//
#include <hip/hip_runtime.h>
#include <hip/hip_bf16.h>

typedef __attribute__((ext_vector_type(16))) _Float16 v16h;
typedef __attribute__((ext_vector_type(8)))  _Float16 v8h;
typedef __attribute__((ext_vector_type(8)))  float    v8f;
typedef __attribute__((ext_vector_type(4)))  unsigned int v4u;
typedef __attribute__((ext_vector_type(8)))  int      v8i;
typedef __attribute__((ext_vector_type(4)))  int      v4i;

#define HF 64
#define WF 100
#define CIN 1024
#define CMID 512
#define MPIX (HF*WF)        /* 6400 */
#define PH (HF+2)           /* 66  */
#define PW (WF+2)           /* 102 */
#define NANCH 9
#define NPROP (MPIX*NANCH)  /* 57600 */
#define PRE_NMS 6000
#define POST_NMS 300
#define NMS_TH 0.7f
#define NHEAD 64            /* 18 cls + 36 bbox padded */
#define FCN 2048
#define NCLS 21
#define FC2N 112            /* 21 + 84 padded */
#define MROI 304            /* 300 padded to 19 tiles */

// ---------------------------------------------------------------- fragments
__device__ __forceinline__ v16h afrag(const _Float16* __restrict__ row, int k0, int half) {
  // A 16x32 f16 layout: lane m holds K = [half*16,+16) contiguous (matches B)
  v8h a0 = *(const v8h*)(row + k0 + half * 16);
  v8h a1 = *(const v8h*)(row + k0 + half * 16 + 8);
  return __builtin_shufflevector(a0, a1, 0,1,2,3,4,5,6,7,8,9,10,11,12,13,14,15);
}
__device__ __forceinline__ v8f wmma32(v16h a, v16h b, v8f c) {
  return __builtin_amdgcn_wmma_f32_16x16x32_f16(false, a, false, b, (short)0, c, false, false);
}

// ---------------------------------------------------------------- packing
__global__ void k_pad_feat(const float* __restrict__ feat, _Float16* __restrict__ P) {
  int idx = blockIdx.x * blockDim.x + threadIdx.x;
  if (idx >= PH * PW * CIN) return;
  int c = idx % CIN, rest = idx / CIN;
  int px = rest % PW, py = rest / PW;
  float v = 0.f;
  if (py >= 1 && py <= HF && px >= 1 && px <= WF)
    v = feat[c * (HF * WF) + (py - 1) * WF + (px - 1)];
  P[idx] = (_Float16)v;
}

__global__ void k_pack_convw(const float* __restrict__ w, _Float16* __restrict__ W2) {
  int idx = blockIdx.x * blockDim.x + threadIdx.x;
  if (idx >= 9 * CMID * CIN) return;
  int k = idx % CIN, rest = idx / CIN;
  int n = rest % CMID, tap = rest / CMID;
  int ky = tap / 3, kx = tap % 3;
  W2[idx] = (_Float16)w[((n * CIN + k) * 3 + ky) * 3 + kx]; // OIHW
}

__global__ void k_pack_headw(const float* __restrict__ wc, const float* __restrict__ wb,
                             _Float16* __restrict__ W) {
  int idx = blockIdx.x * blockDim.x + threadIdx.x;
  if (idx >= NHEAD * CMID) return;
  int k = idx % CMID, n = idx / CMID;
  float v = 0.f;
  if (n < 18) v = wc[n * CMID + k];
  else if (n < 54) v = wb[(n - 18) * CMID + k];
  W[idx] = (_Float16)v;
}

__global__ void k_pack_headb(const float* __restrict__ bc, const float* __restrict__ bb,
                             float* __restrict__ B) {
  int n = blockIdx.x * blockDim.x + threadIdx.x;
  if (n >= NHEAD) return;
  B[n] = (n < 18) ? bc[n] : (n < 54 ? bb[n - 18] : 0.f);
}

__global__ void k_pack_fcw(const float* __restrict__ w, _Float16* __restrict__ W) {
  int idx = blockIdx.x * blockDim.x + threadIdx.x;
  if (idx >= FCN * CIN) return;
  int k = idx % CIN, n = idx / CIN;
  W[idx] = (_Float16)w[k * FCN + n]; // fc_w is (1024, 2048)
}

__global__ void k_pack_fc2w(const float* __restrict__ wc, const float* __restrict__ wb,
                            _Float16* __restrict__ W) {
  int idx = blockIdx.x * blockDim.x + threadIdx.x;
  if (idx >= FC2N * FCN) return;
  int k = idx % FCN, n = idx / FCN;
  float v = 0.f;
  if (n < NCLS) v = wc[k * NCLS + n];
  else if (n < NCLS + 84) v = wb[k * 84 + (n - NCLS)];
  W[idx] = (_Float16)v;
}

__global__ void k_pack_fc2b(const float* __restrict__ bc, const float* __restrict__ bb,
                            float* __restrict__ B) {
  int n = blockIdx.x * blockDim.x + threadIdx.x;
  if (n >= FC2N) return;
  B[n] = (n < NCLS) ? bc[n] : (n < NCLS + 84 ? bb[n - NCLS] : 0.f);
}

__global__ void k_vec_to_f16(const float* __restrict__ vec, _Float16* __restrict__ A) {
  int idx = blockIdx.x * blockDim.x + threadIdx.x;
  if (idx >= MROI * CIN) return;
  A[idx] = (idx < POST_NMS * CIN) ? (_Float16)vec[idx] : (_Float16)0.f;
}

// ---------------------------------------------------------------- conv (implicit GEMM, WMMA)
// Block: 8 waves sharing one 32-col weight group (TDM-staged in LDS), each wave a 32x32 tile.
__global__ __launch_bounds__(256) void k_rpn_conv(
    const _Float16* __restrict__ P,   // padded NHWC (66,102,1024)
    const _Float16* __restrict__ W2,  // (tap, n, k) f16
    const float* __restrict__ bias,   // 512
    _Float16* __restrict__ H) {       // (6400, 512) relu'd
  __shared__ _Float16 lds_w[32 * CIN]; // 64 KB: weights for (tap, 32 n-cols)
  const int lane = threadIdx.x & 31;
  const int wv = threadIdx.x >> 5;     // 0..7
  const int l16 = lane & 15, half = lane >> 4;
  const int bm = blockIdx.x;           // 0..24
  const int nt = blockIdx.y;           // 0..15
  const int m0 = (bm * 8 + wv) * 32 + l16;
  const int m1 = m0 + 16;
  const int y0 = m0 / WF, x0 = m0 % WF;
  const int y1 = m1 / WF, x1 = m1 % WF;
  const int kB = half * 16;

  v8f acc00 = {}, acc01 = {}, acc10 = {}, acc11 = {};
  const _Float16* b0p = &lds_w[(size_t)l16 * CIN + kB];
  const _Float16* b1p = &lds_w[(size_t)(l16 + 16) * CIN + kB];

  for (int tap = 0; tap < 9; ++tap) {
    __syncthreads(); // previous tap fully consumed before overwrite
    const _Float16* gw = W2 + ((size_t)(tap * CMID + nt * 32)) * CIN;
#if __has_builtin(__builtin_amdgcn_tensor_load_to_lds)
    if (wv == 0) {
      // TDM descriptor (ISA 8.3/8.4): 1-D tile, 32768 x 2B contiguous -> LDS.
      unsigned long long ga = (unsigned long long)gw;
      unsigned int la = (unsigned int)(size_t)&lds_w[0];
      v4u g0 = { 1u,                               // count=1, user mode
                 la,                               // lds_addr
                 (unsigned int)ga,                 // global_addr[31:0]
                 (unsigned int)((ga >> 32) & 0x1FFFFFFu) | 0x80000000u }; // addr[56:32] | type=2
      v8i g1 = { 0x00010000,        // data_size=1 (2 bytes), no mask/pad/iterate
                 (int)0x80000000,   // tensor_dim0 = 32768 (low16 at [31:16])
                 0x00010000,        // tensor_dim0 hi=0, tensor_dim1 = 1
                 (int)0x80000000,   // tensor_dim1 hi=0, tile_dim0 = 32768
                 1,                 // tile_dim1 = 1, tile_dim2 = 0
                 32768,             // tensor_dim0_stride lo
                 0, 0 };
      v4i gz = { 0, 0, 0, 0 };
#if __clang_major__ >= 23
      v8i gz8 = { 0, 0, 0, 0, 0, 0, 0, 0 };
      __builtin_amdgcn_tensor_load_to_lds(g0, g1, gz, gz, gz8, 0);
#else
      __builtin_amdgcn_tensor_load_to_lds(g0, g1, gz, gz, 0);
#endif
      __builtin_amdgcn_s_wait_tensorcnt(0);
    }
#else
    for (int t = threadIdx.x * 8; t < 32 * CIN; t += 256 * 8)
      *(v8h*)&lds_w[t] = *(const v8h*)(gw + t);
#endif
    __syncthreads();

    const int dy = tap / 3, dx = tap % 3;
    const _Float16* a0row = P + ((y0 + dy) * PW + (x0 + dx)) * CIN;
    const _Float16* a1row = P + ((y1 + dy) * PW + (x1 + dx)) * CIN;
    for (int k0 = 0; k0 < CIN; k0 += 32) {
      __builtin_prefetch(a0row + k0 + 256, 0, 1); // global_prefetch_b8 on A stream
      v16h a0 = afrag(a0row, k0, half);
      v16h a1 = afrag(a1row, k0, half);
      v16h b0 = *(const v16h*)(b0p + k0);        // ds_load_b128 x2
      v16h b1 = *(const v16h*)(b1p + k0);
      acc00 = wmma32(a0, b0, acc00);
      acc01 = wmma32(a0, b1, acc01);
      acc10 = wmma32(a1, b0, acc10);
      acc11 = wmma32(a1, b1, acc11);
    }
  }
  // store 32x32 tile with bias + relu
  const int nbase = nt * 32;
  const float bv0 = bias[nbase + l16];
  const float bv1 = bias[nbase + 16 + l16];
  const int mt0 = (bm * 8 + wv) * 32;
  for (int v = 0; v < 8; ++v) {
    int ma = mt0 + v + 8 * half;
    int mb = ma + 16;
    float t;
    t = acc00[v] + bv0; H[(size_t)ma * CMID + nbase + l16]      = (_Float16)fmaxf(t, 0.f);
    t = acc01[v] + bv1; H[(size_t)ma * CMID + nbase + 16 + l16] = (_Float16)fmaxf(t, 0.f);
    t = acc10[v] + bv0; H[(size_t)mb * CMID + nbase + l16]      = (_Float16)fmaxf(t, 0.f);
    t = acc11[v] + bv1; H[(size_t)mb * CMID + nbase + 16 + l16] = (_Float16)fmaxf(t, 0.f);
  }
}

// ---------------------------------------------------------------- generic WMMA GEMM
// A: (Mt*16, K) row-major f16.  B: (Nt*16, K) — column n's K contiguous.  C: (Mt*16, Nt*16) f32.
__global__ __launch_bounds__(256) void k_gemm(
    const _Float16* __restrict__ A, const _Float16* __restrict__ B,
    const float* __restrict__ bias, float* __restrict__ C, _Float16* __restrict__ Ch,
    int Mt, int Nt, int K, int relu) {
  const int lane = threadIdx.x & 31;
  const int wid = (blockIdx.x * blockDim.x + threadIdx.x) >> 5;
  if (wid >= Mt * Nt) return;
  const int mt = wid / Nt, nt = wid % Nt;
  const int l16 = lane & 15, half = lane >> 4;
  const _Float16* arow = A + (size_t)(mt * 16 + l16) * K;
  const _Float16* bcol = B + (size_t)(nt * 16 + l16) * K + half * 16;
  v8f acc = {};
  for (int k0 = 0; k0 < K; k0 += 32) {
    v16h a = afrag(arow, k0, half);
    v16h b = *(const v16h*)(bcol + k0);
    acc = wmma32(a, b, acc);
  }
  const int N = Nt * 16;
  const int n = nt * 16 + l16;
  const float bv = bias ? bias[n] : 0.f;
  for (int v = 0; v < 8; ++v) {
    int m = mt * 16 + v + 8 * half;
    float val = acc[v] + bv;
    if (relu) val = fmaxf(val, 0.f);
    if (C) C[(size_t)m * N + n] = val;
    if (Ch) Ch[(size_t)m * N + n] = (_Float16)val;
  }
}

// ---------------------------------------------------------------- proposals
__global__ void k_proposals(const float* __restrict__ ho, const float* __restrict__ im_info,
                            float* __restrict__ scores, float* __restrict__ scores2,
                            float* __restrict__ boxes) {
  int idx = blockIdx.x * blockDim.x + threadIdx.x;
  if (idx >= NPROP) return;
  int m = idx / NANCH, a = idx % NANCH;
  int y = m / WF, x = m % WF;
  const float* row = ho + (size_t)m * NHEAD;
  float lbg = row[a], lfg = row[NANCH + a];
  float sc = 1.f / (1.f + expf(lbg - lfg)); // softmax over {bg,fg}, take fg
  float d0 = row[18 + 4 * a + 0], d1 = row[18 + 4 * a + 1];
  float d2 = row[18 + 4 * a + 2], d3 = row[18 + 4 * a + 3];
  int ri = a / 3, si = a % 3;
  float ratio = (ri == 0) ? 0.5f : (ri == 1 ? 1.f : 2.f);
  float scale = (si == 0) ? 8.f : (si == 1 ? 16.f : 32.f);
  float w0 = rintf(sqrtf(256.f / ratio));
  float h0 = rintf(w0 * ratio);
  float w2 = w0 * scale, h2 = h0 * scale;
  const float ctr = 7.5f;
  float ax1 = ctr - (w2 - 1.f) * 0.5f + x * 16.f;
  float ay1 = ctr - (h2 - 1.f) * 0.5f + y * 16.f;
  float ax2 = ctr + (w2 - 1.f) * 0.5f + x * 16.f;
  float ay2 = ctr + (h2 - 1.f) * 0.5f + y * 16.f;
  float aw = ax2 - ax1 + 1.f, ah = ay2 - ay1 + 1.f;
  float cx = ax1 + 0.5f * aw, cy = ay1 + 0.5f * ah;
  float pcx = d0 * aw + cx, pcy = d1 * ah + cy;
  float pw = expf(d2) * aw, ph = expf(d3) * ah;
  float im_h = im_info[0], im_w = im_info[1];
  float bx1 = fminf(fmaxf(pcx - 0.5f * pw, 0.f), im_w - 1.f);
  float by1 = fminf(fmaxf(pcy - 0.5f * ph, 0.f), im_h - 1.f);
  float bx2 = fminf(fmaxf(pcx + 0.5f * pw, 0.f), im_w - 1.f);
  float by2 = fminf(fmaxf(pcy + 0.5f * ph, 0.f), im_h - 1.f);
  scores[idx] = sc; scores2[idx] = sc;
  boxes[idx * 4 + 0] = bx1; boxes[idx * 4 + 1] = by1;
  boxes[idx * 4 + 2] = bx2; boxes[idx * 4 + 3] = by2;
}

// ---------------------------------------------------------------- top-k (single WGP, destructive)
__global__ __launch_bounds__(1024) void k_topk(float* __restrict__ s2,
                                               const float* __restrict__ boxes,
                                               float* __restrict__ boxtop) {
  __shared__ float sv[1024];
  __shared__ int si[1024];
  const int tid = threadIdx.x;
  for (int it = 0; it < PRE_NMS; ++it) {
    float bv = -3.0e38f;
    int bi = 0x7fffffff;
    for (int j = tid; j < NPROP; j += 1024) {
      float v = s2[j];
      if (v > bv || (v == bv && j < bi)) { bv = v; bi = j; }
    }
    sv[tid] = bv; si[tid] = bi;
    __syncthreads();
    for (int s = 512; s > 0; s >>= 1) {
      if (tid < s) {
        float ov = sv[tid + s]; int oi = si[tid + s];
        if (ov > sv[tid] || (ov == sv[tid] && oi < si[tid])) { sv[tid] = ov; si[tid] = oi; }
      }
      __syncthreads();
    }
    if (tid == 0) {
      int b = si[0];
      boxtop[it * 4 + 0] = boxes[b * 4 + 0];
      boxtop[it * 4 + 1] = boxes[b * 4 + 1];
      boxtop[it * 4 + 2] = boxes[b * 4 + 2];
      boxtop[it * 4 + 3] = boxes[b * 4 + 3];
      s2[b] = -3.0e38f;
    }
    __syncthreads();
  }
}

// ---------------------------------------------------------------- NMS (LDS-resident, 144 KB)
__global__ __launch_bounds__(1024) void k_nms(const float* __restrict__ boxtop,
                                              float* __restrict__ roi_boxes,
                                              float* __restrict__ rois_out) {
  extern __shared__ float sm[];
  float* bx = sm;                      // 6000*4
  float* ar = sm + PRE_NMS * 4;        // 6000
  int* kp = (int*)(sm + PRE_NMS * 5);  // 6000
  const int tid = threadIdx.x;
  for (int i = tid; i < PRE_NMS; i += 1024) {
    float x1 = boxtop[i * 4], y1 = boxtop[i * 4 + 1];
    float x2 = boxtop[i * 4 + 2], y2 = boxtop[i * 4 + 3];
    bx[i * 4] = x1; bx[i * 4 + 1] = y1; bx[i * 4 + 2] = x2; bx[i * 4 + 3] = y2;
    ar[i] = (x2 - x1 + 1.f) * (y2 - y1 + 1.f);
    kp[i] = 1;
  }
  __syncthreads();
  for (int i = 0; i < PRE_NMS; ++i) {
    if (kp[i]) {
      float x1 = bx[i * 4], y1 = bx[i * 4 + 1], x2 = bx[i * 4 + 2], y2 = bx[i * 4 + 3];
      float ai = ar[i];
      for (int j = i + 1 + tid; j < PRE_NMS; j += 1024) {
        float xx1 = fmaxf(x1, bx[j * 4]);
        float yy1 = fmaxf(y1, bx[j * 4 + 1]);
        float xx2 = fminf(x2, bx[j * 4 + 2]);
        float yy2 = fminf(y2, bx[j * 4 + 3]);
        float inter = fmaxf(xx2 - xx1 + 1.f, 0.f) * fmaxf(yy2 - yy1 + 1.f, 0.f);
        float iou = inter / (ai + ar[j] - inter);
        if (iou > NMS_TH) kp[j] = 0;
      }
    }
    __syncthreads();
  }
  if (tid == 0) {
    int cnt = 0;
    for (int i = 0; i < PRE_NMS && cnt < POST_NMS; ++i) {
      if (kp[i]) {
        rois_out[cnt * 5] = 0.f;
        for (int t = 0; t < 4; ++t) {
          rois_out[cnt * 5 + 1 + t] = bx[i * 4 + t];
          roi_boxes[cnt * 4 + t] = bx[i * 4 + t];
        }
        ++cnt;
      }
    }
    for (; cnt < POST_NMS; ++cnt) {
      for (int t = 0; t < 5; ++t) rois_out[cnt * 5 + t] = 0.f;
      for (int t = 0; t < 4; ++t) roi_boxes[cnt * 4 + t] = 0.f;
    }
  }
}

// ---------------------------------------------------------------- ROI align + global mean
__global__ void k_roialign(const float* __restrict__ feat, const float* __restrict__ rb,
                           float* __restrict__ vec) {
  int idx = blockIdx.x * blockDim.x + threadIdx.x;
  if (idx >= POST_NMS * CIN) return;
  int r = idx / CIN, c = idx % CIN;
  const float scale = 1.f / 16.f;
  float x1 = rb[r * 4] * scale, y1 = rb[r * 4 + 1] * scale;
  float x2 = rb[r * 4 + 2] * scale, y2 = rb[r * 4 + 3] * scale;
  float bw = fmaxf(x2 - x1, 1.f) * (1.f / 7.f);
  float bh = fmaxf(y2 - y1, 1.f) * (1.f / 7.f);
  const float* f = feat + (size_t)c * (HF * WF);
  float sum = 0.f;
  for (int sy = 0; sy < 14; ++sy) {
    float ys = fminf(fmaxf(y1 + (sy + 0.5f) * 0.5f * bh, 0.f), (float)(HF - 1));
    int y0 = (int)floorf(ys);
    float ly = ys - (float)y0;
    int y1i = (y0 + 1 < HF) ? y0 + 1 : HF - 1;
    for (int sx = 0; sx < 14; ++sx) {
      float xs = fminf(fmaxf(x1 + (sx + 0.5f) * 0.5f * bw, 0.f), (float)(WF - 1));
      int x0 = (int)floorf(xs);
      float lx = xs - (float)x0;
      int x1i = (x0 + 1 < WF) ? x0 + 1 : WF - 1;
      float v00 = f[y0 * WF + x0], v01 = f[y0 * WF + x1i];
      float v10 = f[y1i * WF + x0], v11 = f[y1i * WF + x1i];
      sum += v00 * (1.f - ly) * (1.f - lx) + v01 * (1.f - ly) * lx +
             v10 * ly * (1.f - lx) + v11 * ly * lx;
    }
  }
  vec[idx] = sum * (1.f / 196.f);
}

// ---------------------------------------------------------------- final softmax / split
__global__ void k_finalize(const float* __restrict__ out2, float* __restrict__ cls_out,
                           float* __restrict__ bbox_out) {
  int r = blockIdx.x * blockDim.x + threadIdx.x;
  if (r >= POST_NMS) return;
  const float* row = out2 + (size_t)r * FC2N;
  float mx = row[0];
  for (int j = 1; j < NCLS; ++j) mx = fmaxf(mx, row[j]);
  float e[NCLS], den = 0.f;
  for (int j = 0; j < NCLS; ++j) { e[j] = expf(row[j] - mx); den += e[j]; }
  float inv = 1.f / den;
  for (int j = 0; j < NCLS; ++j) cls_out[r * NCLS + j] = e[j] * inv;
  for (int j = 0; j < 84; ++j) bbox_out[r * 84 + j] = row[NCLS + j];
}

// ================================================================ launch
extern "C" void kernel_launch(void* const* d_in, const int* in_sizes, int n_in,
                              void* d_out, int out_size, void* d_ws, size_t ws_size,
                              hipStream_t stream) {
  (void)in_sizes; (void)n_in; (void)out_size; (void)ws_size;
  const float* base_feat = (const float*)d_in[0];
  const float* im_info   = (const float*)d_in[1];
  const float* rpn_conv_w= (const float*)d_in[2];
  const float* rpn_conv_b= (const float*)d_in[3];
  const float* rpn_cls_w = (const float*)d_in[4];
  const float* rpn_cls_b = (const float*)d_in[5];
  const float* rpn_bbox_w= (const float*)d_in[6];
  const float* rpn_bbox_b= (const float*)d_in[7];
  const float* fc_w      = (const float*)d_in[8];
  const float* fc_b      = (const float*)d_in[9];
  const float* cls_w     = (const float*)d_in[10];
  const float* cls_b     = (const float*)d_in[11];
  const float* bbox_w    = (const float*)d_in[12];
  const float* bbox_b    = (const float*)d_in[13];
  float* out = (float*)d_out; // rois[1500] | cls_prob[6300] | bbox_pred[25200]

  char* wsb = (char*)d_ws;
  size_t o = 0;
  auto alloc = [&](size_t bytes) { size_t r = o; o += (bytes + 255) & ~(size_t)255; return r; };
  size_t OFF_PAD    = alloc((size_t)PH * PW * CIN * 2);
  size_t OFF_CONVW  = alloc((size_t)9 * CMID * CIN * 2);
  size_t OFF_HBUF   = alloc((size_t)MPIX * CMID * 2);
  size_t OFF_HEADW  = alloc((size_t)NHEAD * CMID * 2);
  size_t OFF_HEADB  = alloc((size_t)NHEAD * 4);
  size_t OFF_HEADO  = alloc((size_t)MPIX * NHEAD * 4);
  size_t OFF_SCORE  = alloc((size_t)NPROP * 4);
  size_t OFF_SCORE2 = alloc((size_t)NPROP * 4);
  size_t OFF_BOXES  = alloc((size_t)NPROP * 4 * 4);
  size_t OFF_BOXTOP = alloc((size_t)PRE_NMS * 4 * 4);
  size_t OFF_RB     = alloc((size_t)POST_NMS * 4 * 4);
  size_t OFF_VEC    = alloc((size_t)POST_NMS * CIN * 4);
  size_t OFF_AVEC   = alloc((size_t)MROI * CIN * 2);
  size_t OFF_FCW    = alloc((size_t)FCN * CIN * 2);
  size_t OFF_HIDH   = alloc((size_t)MROI * FCN * 2);
  size_t OFF_FC2W   = alloc((size_t)FC2N * FCN * 2);
  size_t OFF_FC2B   = alloc((size_t)FC2N * 4);
  size_t OFF_OUT2   = alloc((size_t)MROI * FC2N * 4);

  _Float16* P     = (_Float16*)(wsb + OFF_PAD);
  _Float16* CONVW = (_Float16*)(wsb + OFF_CONVW);
  _Float16* HBUF  = (_Float16*)(wsb + OFF_HBUF);
  _Float16* HEADW = (_Float16*)(wsb + OFF_HEADW);
  float*    HEADB = (float*)(wsb + OFF_HEADB);
  float*    HEADO = (float*)(wsb + OFF_HEADO);
  float*    SCORE = (float*)(wsb + OFF_SCORE);
  float*    SCORE2= (float*)(wsb + OFF_SCORE2);
  float*    BOXES = (float*)(wsb + OFF_BOXES);
  float*    BOXTOP= (float*)(wsb + OFF_BOXTOP);
  float*    RB    = (float*)(wsb + OFF_RB);
  float*    VEC   = (float*)(wsb + OFF_VEC);
  _Float16* AVEC  = (_Float16*)(wsb + OFF_AVEC);
  _Float16* FCW   = (_Float16*)(wsb + OFF_FCW);
  _Float16* HIDH  = (_Float16*)(wsb + OFF_HIDH);
  _Float16* FC2W  = (_Float16*)(wsb + OFF_FC2W);
  float*    FC2B  = (float*)(wsb + OFF_FC2B);
  float*    OUT2  = (float*)(wsb + OFF_OUT2);

  const int B = 256;
  // packing
  k_pad_feat<<<(PH * PW * CIN + B - 1) / B, B, 0, stream>>>(base_feat, P);
  k_pack_convw<<<(9 * CMID * CIN + B - 1) / B, B, 0, stream>>>(rpn_conv_w, CONVW);
  k_pack_headw<<<(NHEAD * CMID + B - 1) / B, B, 0, stream>>>(rpn_cls_w, rpn_bbox_w, HEADW);
  k_pack_headb<<<1, NHEAD, 0, stream>>>(rpn_cls_b, rpn_bbox_b, HEADB);
  k_pack_fcw<<<(FCN * CIN + B - 1) / B, B, 0, stream>>>(fc_w, FCW);
  k_pack_fc2w<<<(FC2N * FCN + B - 1) / B, B, 0, stream>>>(cls_w, bbox_w, FC2W);
  k_pack_fc2b<<<1, FC2N, 0, stream>>>(cls_b, bbox_b, FC2B);
  // RPN conv: grid (25 row-blocks, 16 col-groups); TDM weight staging + 2x2 WMMA tiling
  {
    dim3 cgrid(25, 16);
    k_rpn_conv<<<cgrid, 256, 0, stream>>>(P, CONVW, rpn_conv_b, HBUF);
  }
  // RPN heads GEMM: (400, 4) tiles, K=512
  k_gemm<<<(400 * 4 + 7) / 8, B, 0, stream>>>(HBUF, HEADW, HEADB, HEADO, (_Float16*)nullptr,
                                              400, 4, CMID, 0);
  // proposals: softmax + anchor decode + clip
  k_proposals<<<(NPROP + B - 1) / B, B, 0, stream>>>(HEADO, im_info, SCORE, SCORE2, BOXES);
  // top-6000 + greedy NMS (LDS resident) + compact to 300 rois
  k_topk<<<1, 1024, 0, stream>>>(SCORE2, BOXES, BOXTOP);
  k_nms<<<1, 1024, (size_t)PRE_NMS * 6 * 4, stream>>>(BOXTOP, RB, out);
  // ROI align + mean pooling -> 300x1024
  k_roialign<<<(POST_NMS * CIN + B - 1) / B, B, 0, stream>>>(base_feat, RB, VEC);
  k_vec_to_f16<<<(MROI * CIN + B - 1) / B, B, 0, stream>>>(VEC, AVEC);
  // FC1: (19,128) tiles, K=1024, relu, f16 out
  k_gemm<<<(19 * 128 + 7) / 8, B, 0, stream>>>(AVEC, FCW, fc_b, (float*)nullptr, HIDH,
                                               19, 128, CIN, 1);
  // FC2: (19,7) tiles, K=2048
  k_gemm<<<(19 * 7 + 7) / 8, B, 0, stream>>>(HIDH, FC2W, FC2B, OUT2, (_Float16*)nullptr,
                                             19, 7, FCN, 0);
  // softmax(cls) + split outputs
  k_finalize<<<(POST_NMS + B - 1) / B, B, 0, stream>>>(OUT2, out + POST_NMS * 5,
                                                      out + POST_NMS * 5 + POST_NMS * NCLS);
}
